// EntanglementPropagator_15384572854798
// MI455X (gfx1250) — matrix operand order repaired and checked by reading, not compiled, not run on hardware
//
#include <hip/hip_runtime.h>
#include <hip/hip_bf16.h>

typedef __attribute__((ext_vector_type(16))) _Float16 v16h;
typedef __attribute__((ext_vector_type(8)))  _Float16 v8h;
typedef __attribute__((ext_vector_type(8)))  float    v8f;

#define Nn 256        // nodes
#define Ff 256        // features
#define Bb 32         // batch
#define JT 64         // dst-column tile per block
#define GSSTR 65      // Gs f32 row stride (64 cols + 1 pad)
#define GTSTR 272     // Gt f16 row stride (256 + 16 pad; 17*16 halves -> 32B-aligned rows, bank step 8)
#define NFSTR 264     // nfs f16 row stride (256 + 8 pad; 33*8 halves -> 16B-aligned rows, bank step 4)

// LDS layout (bytes):
#define GS_BYTES  (Nn * GSSTR * 4)            //  66560 : Gs[i][jl] f32 scatter buffer
#define GT_BYTES  (JT * GTSTR * 2)            //  34816 : Gt[jl][i] f16 transposed (B operand)
#define NFS_BYTES (Bb * NFSTR * 2)            //  16896 : nfs[b][i] f16 (A operand)
#define CNT_BYTES (JT * 4)                    //    256 : per-tile source counts
#define SMEM_BYTES (GS_BYTES + GT_BYTES + NFS_BYTES + CNT_BYTES)   // 118528 -> 2 blocks/WGP

__global__ __launch_bounds__(256)
void EntanglementPropagator_15384572854798_kernel(
        const float* __restrict__ nf,     // [B, N, F]
        const float* __restrict__ W,      // [N, N, F]
        const float* __restrict__ phase,  // [N, N]
        const int*   __restrict__ src,    // [E]
        const int*   __restrict__ dst,    // [E]
        float*       __restrict__ out,    // [B, N, F]
        int E)
{
    extern __shared__ char smem[];
    float*    Gs  = (float*)smem;
    _Float16* Gt  = (_Float16*)(smem + GS_BYTES);
    _Float16* nfs = (_Float16*)(smem + GS_BYTES + GT_BYTES);
    int*      cnt = (int*)(smem + GS_BYTES + GT_BYTES + NFS_BYTES);

    const int tid = threadIdx.x;
    const int f   = blockIdx.x >> 2;          // feature index (256)
    const int jb  = (blockIdx.x & 3) * JT;    // dst-column tile base (4 tiles of 64)

    // ---- 1) zero Gs / cnt, load nf[:, :, f] slice -> f16 (b-major) ----
    for (int u = tid; u < Nn * GSSTR; u += 256) Gs[u] = 0.0f;
    if (tid < JT) cnt[tid] = 0;
    for (int u = tid; u < Bb * Nn; u += 256) {
        int b = u >> 8, i = u & 255;
        nfs[b * NFSTR + i] = (_Float16)nf[((b << 8) + i) * Ff + f];
    }
    __syncthreads();

    // ---- 2) edge scan: tile-local degree counts + scatter eff. weights ----
    for (int e = tid; e < E; e += 256) {
        int s  = src[e];
        int d  = dst[e];
        int sl = s - jb;
        if ((unsigned)sl < (unsigned)JT) atomicAdd(&cnt[sl], 1);   // ds_add_u32
        int jl = d - jb;
        if ((unsigned)jl < (unsigned)JT) {
            int p = (s << 8) + d;                                  // src*N + dst
            float v = W[p * Ff + f] * cosf(phase[p]);
            atomicAdd(&Gs[s * GSSTR + jl], v);                     // ds_add_f32
        }
    }
    __syncthreads();

    // ---- 3) convert + transpose: Gt[j][i] = (f16)Gs[i][j], packed stores ----
    for (int u = tid; u < JT * (Nn / 2); u += 256) {
        int j  = u >> 7;          // 0..63
        int i2 = u & 127;         // pair index: i = 2*i2, 2*i2+1
        union { unsigned u32; _Float16 h[2]; } pk;
        pk.h[0] = (_Float16)Gs[(2 * i2)     * GSSTR + j];
        pk.h[1] = (_Float16)Gs[(2 * i2 + 1) * GSSTR + j];
        *(unsigned*)(Gt + j * GTSTR + 2 * i2) = pk.u32;            // ds_store_b32
    }
    __syncthreads();

    // ---- 4) WMMA: out[:, jb:jb+64, f] = nf_slice(32x256) @ G(256x64) ----
    const int lane = tid & 31;
    const int l    = lane & 15;
    const int hi   = lane >> 4;
    const int wave = tid >> 5;
    const int mt   = wave >> 2;          // batch-row tile (2)
    const int nt   = wave & 3;           // col tile (4)

    const int brow = (mt << 4) + l;      // A row (= batch index) for this lane
    const int jc   = (nt << 4) + l;      // local dst col for this lane

    const _Float16* arow = nfs + brow * NFSTR;   // 528B stride: 16B-aligned
    const _Float16* bcol = Gt  + jc   * GTSTR;   // 544B stride: 32B-aligned

    v8f c = {};
    #pragma unroll
    for (int k0 = 0; k0 < Nn; k0 += 32) {
        // A fragment (16-bit 16x32): halves 0-7 -> K = k0+hi*8+h ; 8-15 -> K = k0+16+hi*8+h
        v8h alo = *(const v8h*)(arow + k0 + (hi << 3));
        v8h ahi = *(const v8h*)(arow + k0 + 16 + (hi << 3));
        v16h a  = __builtin_shufflevector(alo, ahi,
                    0,1,2,3,4,5,6,7,8,9,10,11,12,13,14,15);
        // B fragment (32x16): halves 0-7 -> K = k0+hi*8+h ; 8-15 -> K = k0+16+hi*8+h, N = l
        v8h blo = *(const v8h*)(bcol + k0 + (hi << 3));
        v8h bhi = *(const v8h*)(bcol + k0 + 16 + (hi << 3));
        v16h b  = __builtin_shufflevector(blo, bhi,
                    0,1,2,3,4,5,6,7,8,9,10,11,12,13,14,15);
        c = __builtin_amdgcn_wmma_f32_16x16x32_f16(false, a, false, b, (short)0, c, false, false);
    }

    // ---- 5) epilogue: scale by 1/max(outdeg,1) via v_rcp_f32, store ----
    float n = (float)cnt[jc];
    n = n > 1.0f ? n : 1.0f;
    const float scale = __builtin_amdgcn_rcpf(n);

    const int jg    = jb + jc;                 // global dst column
    const int bbase = (mt << 4) + (hi << 3);   // C/D layout: VGPR r -> M = r + 8*hi
    #pragma unroll
    for (int r = 0; r < 8; ++r) {
        int b = bbase + r;
        out[(((b << 8) + jg) << 8) + f] = c[r] * scale;
    }
}

extern "C" void kernel_launch(void* const* d_in, const int* in_sizes, int n_in,
                              void* d_out, int out_size, void* d_ws, size_t ws_size,
                              hipStream_t stream) {
    const float* nf    = (const float*)d_in[0];
    const float* W     = (const float*)d_in[1];
    const float* phase = (const float*)d_in[2];
    const int*   src   = (const int*)d_in[3];
    const int*   dst   = (const int*)d_in[4];
    float* out = (float*)d_out;
    const int E = in_sizes[3];

    dim3 grid(Ff * (Nn / JT));   // 256 features x 4 column tiles = 1024 blocks
    dim3 block(256);             // 8 waves
    EntanglementPropagator_15384572854798_kernel<<<grid, block, SMEM_BYTES, stream>>>(
        nf, W, phase, src, dst, out, E);

    (void)n_in; (void)out_size; (void)d_ws; (void)ws_size;
}
